// MetaGRUNoShareW_45492293599378
// MI455X (gfx1250) — compile-verified, matrix operand order checked by reading, not run on hardware
//
#include <hip/hip_runtime.h>
#include <hip/hip_bf16.h>

// ---------------------------------------------------------------------------
// MetaLayer GNN (edge/node/global MLP+GRU, 3 shared-weight steps) for gfx1250.
// GEMMs via v_wmma_f32_16x16x32_f16 (f16 operands, f32 accumulate).
// Each wave owns a 32-row tile (2 WMMA M-tiles) so every weight B-fragment
// feeds 2 WMMAs; B-fragments are prefetch-rotated to hide load latency.
// ---------------------------------------------------------------------------

#define H 128
#define NN 50000
#define EE 500000
#define BB 64
#define STEPS 3

typedef __attribute__((ext_vector_type(16))) _Float16 v16h;
typedef __attribute__((ext_vector_type(8)))  _Float16 v8h;
typedef __attribute__((ext_vector_type(8)))  float    v8f;
typedef __attribute__((ext_vector_type(4)))  float    v4f;

__device__ __forceinline__ v8f wmma16(v16h a, v16h b, v8f c) {
  return __builtin_amdgcn_wmma_f32_16x16x32_f16(
      false, a, false, b, (short)0, c, false, false);
}

// Fragment from f16 memory (weights in global, tiles in LDS).
// ISA A/B 16-bit layout: elems 0..7 = K{+0..7}, elems 8..15 = K{+16..23}.
__device__ __forceinline__ v16h frag_f16(const _Float16* p) {
  v8h lo = *(const v8h*)p;
  v8h hi = *(const v8h*)(p + 16);
  v16h r;
#pragma unroll
  for (int i = 0; i < 8; ++i) { r[i] = lo[i]; r[i + 8] = hi[i]; }
  return r;
}

// A fragment straight from f32 global rows (gather path), with per-row scale.
__device__ __forceinline__ v16h frag_f32(const float* p, float scale) {
  v4f a0 = *(const v4f*)p;
  v4f a1 = *(const v4f*)(p + 4);
  v4f b0 = *(const v4f*)(p + 16);
  v4f b1 = *(const v4f*)(p + 20);
  v16h r;
#pragma unroll
  for (int i = 0; i < 4; ++i) {
    r[i]      = (_Float16)(a0[i] * scale);
    r[4 + i]  = (_Float16)(a1[i] * scale);
    r[8 + i]  = (_Float16)(b0[i] * scale);
    r[12 + i] = (_Float16)(b1[i] * scale);
  }
  return r;
}

__device__ __forceinline__ float sigmoidf_(float x) {
  return 1.0f / (1.0f + __expf(-x));
}

// ---------------------------------------------------------------------------
// Fused 32-row tile engine: out = GRU(MLP(concat(segs)), h_old)
//   NSEG 128-wide input segments (K1 = NSEG*128); hidden H=128; gates 3H.
//   segp0/segsc0: per-lane row pointers/scales for rows 0..15 (row = lane&15)
//   segp1/segsc1: same for rows 16..31
//   sA: LDS 32x128 f16 (h1, then MLP output), sH: LDS 32x128 f16 (h_old).
// ---------------------------------------------------------------------------
template <int NSEG>
__device__ __forceinline__ void tile_mlp_gru(
    int lane, _Float16* sA, _Float16* sH,
    const float* const* segp0, const float* segsc0,
    const float* const* segp1, const float* segsc1,
    const _Float16* __restrict__ w1, const float* __restrict__ b1,
    const _Float16* __restrict__ w2, const float* __restrict__ b2,
    const _Float16* __restrict__ wih, const _Float16* __restrict__ whh,
    const float* __restrict__ bih, const float* __restrict__ bhh,
    const float* __restrict__ holdBase,  // tile base, row stride H
    float* __restrict__ outBase,         // tile base, row stride H
    float* __restrict__ out2Base, int out2Stride,
    int rowsValid) {
  const int lm = lane & 15;
  const int h8 = (lane >> 4) << 3;      // K half-offset {0,8}
  const int rowOff = (lane >> 4) << 3;  // C-layout row offset {0,8}
  constexpr int K1 = NSEG * H;

  // Stage h_old tile as f16 (GRU A-operand for W_hh GEMMs); clamp tail reads.
  for (int idx = lane; idx < 32 * H; idx += 32) {
    int r = idx >> 7, c = idx & (H - 1);
    int rc = (r < rowsValid) ? r : (rowsValid - 1);
    sH[idx] = (_Float16)holdBase[(size_t)rc * H + c];
  }

  // ---- MLP layer 1: [32, K1] @ W1^T -> relu -> sA ----
  v8f acc[2][8];
#pragma unroll
  for (int mt = 0; mt < 2; ++mt)
#pragma unroll
    for (int nt = 0; nt < 8; ++nt)
#pragma unroll
      for (int j = 0; j < 8; ++j) acc[mt][nt][j] = 0.0f;

#pragma unroll
  for (int kp = 0; kp < NSEG * 4; ++kp) {
    const int seg = kp >> 2;
    const int c0 = (kp & 3) << 5;
    v16h a0 = frag_f32(segp0[seg] + c0 + h8, segsc0[seg]);
    v16h a1 = frag_f32(segp1[seg] + c0 + h8, segsc1[seg]);
    v16h b = frag_f16(w1 + (size_t)lm * K1 + (kp << 5) + h8);
#pragma unroll
    for (int nt = 0; nt < 8; ++nt) {
      v16h bn = (nt < 7)
          ? frag_f16(w1 + (size_t)((nt + 1) * 16 + lm) * K1 + (kp << 5) + h8)
          : b;  // prefetch next N-tile weights before consuming current
      acc[0][nt] = wmma16(a0, b, acc[0][nt]);
      acc[1][nt] = wmma16(a1, b, acc[1][nt]);
      b = bn;
    }
  }
#pragma unroll
  for (int nt = 0; nt < 8; ++nt) {
    const int col = nt * 16 + lm;
    const float bias = b1[col];
#pragma unroll
    for (int mt = 0; mt < 2; ++mt)
#pragma unroll
      for (int j = 0; j < 8; ++j) {
        float v = acc[mt][nt][j] + bias;
        v = v > 0.0f ? v : 0.0f;
        sA[(mt * 16 + j + rowOff) * H + col] = (_Float16)v;
      }
  }

  // ---- MLP layer 2: [32,128] @ W2^T ----
  v8f acc2[2][8];
#pragma unroll
  for (int mt = 0; mt < 2; ++mt)
#pragma unroll
    for (int nt = 0; nt < 8; ++nt)
#pragma unroll
      for (int j = 0; j < 8; ++j) acc2[mt][nt][j] = 0.0f;
#pragma unroll
  for (int kp = 0; kp < 4; ++kp) {
    const int ko = (kp << 5) + h8;
    v16h a0 = frag_f16(sA + lm * H + ko);
    v16h a1 = frag_f16(sA + (16 + lm) * H + ko);
    v16h b = frag_f16(w2 + (size_t)lm * H + ko);
#pragma unroll
    for (int nt = 0; nt < 8; ++nt) {
      v16h bn = (nt < 7)
          ? frag_f16(w2 + (size_t)((nt + 1) * 16 + lm) * H + ko)
          : b;
      acc2[0][nt] = wmma16(a0, b, acc2[0][nt]);
      acc2[1][nt] = wmma16(a1, b, acc2[1][nt]);
      b = bn;
    }
  }
  // MLP output (+b2) overwrites sA (per-wave in-order LDS keeps this safe)
#pragma unroll
  for (int nt = 0; nt < 8; ++nt) {
    const int col = nt * 16 + lm;
    const float bias = b2[col];
#pragma unroll
    for (int mt = 0; mt < 2; ++mt)
#pragma unroll
      for (int j = 0; j < 8; ++j)
        sA[(mt * 16 + j + rowOff) * H + col] =
            (_Float16)(acc2[mt][nt][j] + bias);
  }

  // ---- GRU: per 16-column gate chunk, fused r/z/n with 12 WMMAs per panel ----
  for (int nt = 0; nt < 8; ++nt) {
    const int col = nt * 16 + lm;
    v8f aR[2], aZ[2], aG[2], aHG[2];
#pragma unroll
    for (int mt = 0; mt < 2; ++mt)
#pragma unroll
      for (int j = 0; j < 8; ++j) {
        aR[mt][j] = 0.f; aZ[mt][j] = 0.f; aG[mt][j] = 0.f; aHG[mt][j] = 0.f;
      }
#pragma unroll
    for (int kp = 0; kp < 4; ++kp) {
      const int ko = (kp << 5) + h8;
      v16h ae0 = frag_f16(sA + lm * H + ko);        // MLP output rows 0..15
      v16h ae1 = frag_f16(sA + (16 + lm) * H + ko); // rows 16..31
      v16h ah0 = frag_f16(sH + lm * H + ko);        // h_old
      v16h ah1 = frag_f16(sH + (16 + lm) * H + ko);
      // issue all six gate-weight fragments up front (one load clause)
      v16h bR1 = frag_f16(wih + (size_t)(0 * H + col) * H + ko);
      v16h bR2 = frag_f16(whh + (size_t)(0 * H + col) * H + ko);
      v16h bZ1 = frag_f16(wih + (size_t)(1 * H + col) * H + ko);
      v16h bZ2 = frag_f16(whh + (size_t)(1 * H + col) * H + ko);
      v16h bG  = frag_f16(wih + (size_t)(2 * H + col) * H + ko);
      v16h bHG = frag_f16(whh + (size_t)(2 * H + col) * H + ko);
      aR[0]  = wmma16(ae0, bR1, aR[0]);  aR[1]  = wmma16(ae1, bR1, aR[1]);
      aR[0]  = wmma16(ah0, bR2, aR[0]);  aR[1]  = wmma16(ah1, bR2, aR[1]);
      aZ[0]  = wmma16(ae0, bZ1, aZ[0]);  aZ[1]  = wmma16(ae1, bZ1, aZ[1]);
      aZ[0]  = wmma16(ah0, bZ2, aZ[0]);  aZ[1]  = wmma16(ah1, bZ2, aZ[1]);
      aG[0]  = wmma16(ae0, bG,  aG[0]);  aG[1]  = wmma16(ae1, bG,  aG[1]);
      aHG[0] = wmma16(ah0, bHG, aHG[0]); aHG[1] = wmma16(ah1, bHG, aHG[1]);
    }
    const float brz = bih[col] + bhh[col];
    const float bzz = bih[H + col] + bhh[H + col];
    const float big = bih[2 * H + col];
    const float bhg = bhh[2 * H + col];
#pragma unroll
    for (int mt = 0; mt < 2; ++mt)
#pragma unroll
      for (int j = 0; j < 8; ++j) {
        const int rr = mt * 16 + j + rowOff;
        if (rr < rowsValid) {
          float rg = sigmoidf_(aR[mt][j] + brz);
          float zg = sigmoidf_(aZ[mt][j] + bzz);
          float ng = tanhf(aG[mt][j] + big + rg * (aHG[mt][j] + bhg));
          float hold = holdBase[(size_t)rr * H + col];
          float o = (1.0f - zg) * ng + zg * hold;
          outBase[(size_t)rr * H + col] = o;
          if (out2Base) out2Base[(size_t)rr * out2Stride + col] = o;
        }
      }
  }
}

// ---------------------------------------------------------------------------
// Stage kernels: 64 threads = 2 waves, 32 rows per wave.
// ---------------------------------------------------------------------------
__global__ void __launch_bounds__(64) edge_kernel(
    const float* __restrict__ x, const int* __restrict__ srcI,
    const int* __restrict__ dstI, const int* __restrict__ batch,
    const float* __restrict__ ea_in, float* __restrict__ ea_out,
    const float* __restrict__ u,
    const _Float16* __restrict__ w1, const float* __restrict__ b1,
    const _Float16* __restrict__ w2, const float* __restrict__ b2,
    const _Float16* __restrict__ wih, const _Float16* __restrict__ whh,
    const float* __restrict__ bih, const float* __restrict__ bhh,
    int numTiles) {
  __shared__ __align__(16) _Float16 sAa[2][32 * H];
  __shared__ __align__(16) _Float16 sHa[2][32 * H];
  const int wv = threadIdx.x >> 5;
  const int lane = threadIdx.x & 31;
  const int tile = blockIdx.x * 2 + wv;
  if (tile >= numTiles) return;  // wave-uniform
  const int e0 = tile * 32;
  const int lm = lane & 15;
  const int eL0 = e0 + lm, eL1 = e0 + 16 + lm;
  const int s0 = srcI[eL0], d0 = dstI[eL0];
  const int s1 = srcI[eL1], d1 = dstI[eL1];
  const float* segp0[4] = {x + (size_t)s0 * H, x + (size_t)d0 * H,
                           ea_in + (size_t)eL0 * H, u + (size_t)batch[s0] * H};
  const float* segp1[4] = {x + (size_t)s1 * H, x + (size_t)d1 * H,
                           ea_in + (size_t)eL1 * H, u + (size_t)batch[s1] * H};
  const float one4[4] = {1.f, 1.f, 1.f, 1.f};
  tile_mlp_gru<4>(lane, sAa[wv], sHa[wv], segp0, one4, segp1, one4,
                  w1, b1, w2, b2, wih, whh, bih, bhh,
                  ea_in + (size_t)e0 * H, ea_out + (size_t)e0 * H,
                  nullptr, 0, 32);
}

__global__ void __launch_bounds__(64) node_kernel(
    const float* __restrict__ x_in, float* __restrict__ x_out,
    const float* __restrict__ agg, const float* __restrict__ acnt,
    const float* __restrict__ u, const int* __restrict__ batch,
    const _Float16* __restrict__ w1, const float* __restrict__ b1,
    const _Float16* __restrict__ w2, const float* __restrict__ b2,
    const _Float16* __restrict__ wih, const _Float16* __restrict__ whh,
    const float* __restrict__ bih, const float* __restrict__ bhh,
    int numTiles) {
  __shared__ __align__(16) _Float16 sAa[2][32 * H];
  __shared__ __align__(16) _Float16 sHa[2][32 * H];
  const int wv = threadIdx.x >> 5;
  const int lane = threadIdx.x & 31;
  const int tile = blockIdx.x * 2 + wv;
  if (tile >= numTiles) return;
  const int n0 = tile * 32;
  const int lm = lane & 15;
  int nL0 = n0 + lm;       if (nL0 > NN - 1) nL0 = NN - 1;  // clamp tail reads
  int nL1 = n0 + 16 + lm;  if (nL1 > NN - 1) nL1 = NN - 1;
  const float sc0 = 1.0f / fmaxf(acnt[nL0], 1.0f);  // fold segment-mean divide
  const float sc1 = 1.0f / fmaxf(acnt[nL1], 1.0f);
  const float* segp0[3] = {x_in + (size_t)nL0 * H, agg + (size_t)nL0 * H,
                           u + (size_t)batch[nL0] * H};
  const float* segp1[3] = {x_in + (size_t)nL1 * H, agg + (size_t)nL1 * H,
                           u + (size_t)batch[nL1] * H};
  const float ssc0[3] = {1.f, sc0, 1.f};
  const float ssc1[3] = {1.f, sc1, 1.f};
  int rowsValid = NN - n0;
  if (rowsValid > 32) rowsValid = 32;
  tile_mlp_gru<3>(lane, sAa[wv], sHa[wv], segp0, ssc0, segp1, ssc1,
                  w1, b1, w2, b2, wih, whh, bih, bhh,
                  x_in + (size_t)n0 * H, x_out + (size_t)n0 * H,
                  nullptr, 0, rowsValid);
}

__global__ void __launch_bounds__(64) global_kernel(
    const float* __restrict__ u_in, float* __restrict__ u_out,
    const float* __restrict__ gm, const float* __restrict__ gcnt,
    const _Float16* __restrict__ w1, const float* __restrict__ b1,
    const _Float16* __restrict__ w2, const float* __restrict__ b2,
    const _Float16* __restrict__ wih, const _Float16* __restrict__ whh,
    const float* __restrict__ bih, const float* __restrict__ bhh,
    float* __restrict__ outp, int step) {
  __shared__ __align__(16) _Float16 sAa[2][32 * H];
  __shared__ __align__(16) _Float16 sHa[2][32 * H];
  const int wv = threadIdx.x >> 5;
  const int lane = threadIdx.x & 31;
  const int b0 = wv * 32;  // B=64 -> 2 waves x 32 rows, one block
  const int lm = lane & 15;
  const int bL0 = b0 + lm, bL1 = b0 + 16 + lm;
  const float sc0 = 1.0f / fmaxf(gcnt[bL0], 1.0f);
  const float sc1 = 1.0f / fmaxf(gcnt[bL1], 1.0f);
  const float* segp0[2] = {u_in + (size_t)bL0 * H, gm + (size_t)bL0 * H};
  const float* segp1[2] = {u_in + (size_t)bL1 * H, gm + (size_t)bL1 * H};
  const float ssc0[2] = {1.f, sc0};
  const float ssc1[2] = {1.f, sc1};
  tile_mlp_gru<2>(lane, sAa[wv], sHa[wv], segp0, ssc0, segp1, ssc1,
                  w1, b1, w2, b2, wih, whh, bih, bhh,
                  u_in + (size_t)b0 * H, u_out + (size_t)b0 * H,
                  outp + ((size_t)b0 * STEPS + step) * H, STEPS * H, 32);
}

// ---------------------------------------------------------------------------
// Support kernels
// ---------------------------------------------------------------------------
__global__ void f32_to_f16_kernel(const float* __restrict__ s,
                                  _Float16* __restrict__ d, int n) {
  int i = blockIdx.x * 256 + threadIdx.x;
  if (i < n) d[i] = (_Float16)s[i];
}

__global__ void zero_kernel(float* __restrict__ p, size_t n) {
  size_t i = (size_t)blockIdx.x * 256 + threadIdx.x;
  if (i < n) p[i] = 0.0f;
}

__global__ void count_kernel(const int* __restrict__ idx, float* __restrict__ cnt,
                             int n) {
  int i = blockIdx.x * 256 + threadIdx.x;
  if (i < n) atomicAdd(&cnt[idx[i]], 1.0f);
}

__global__ void scatter_edges_kernel(const float* __restrict__ ea,
                                     const int* __restrict__ dstI,
                                     float* __restrict__ agg) {
  size_t i = (size_t)blockIdx.x * 256 + threadIdx.x;
  if (i >= (size_t)EE * H) return;
  int e = (int)(i >> 7), c = (int)(i & (H - 1));
  atomicAdd(&agg[(size_t)dstI[e] * H + c], ea[i]);
}

__global__ void scatter_nodes_kernel(const float* __restrict__ x,
                                     const int* __restrict__ batch,
                                     float* __restrict__ gm) {
  size_t i = (size_t)blockIdx.x * 256 + threadIdx.x;
  if (i >= (size_t)NN * H) return;
  int n = (int)(i >> 7), c = (int)(i & (H - 1));
  atomicAdd(&gm[(size_t)batch[n] * H + c], x[i]);
}

// ---------------------------------------------------------------------------
extern "C" void kernel_launch(void* const* d_in, const int* in_sizes, int n_in,
                              void* d_out, int out_size, void* d_ws,
                              size_t ws_size, hipStream_t stream) {
  (void)in_sizes; (void)n_in; (void)out_size; (void)ws_size;
  const float* x0  = (const float*)d_in[0];
  const int* eidx  = (const int*)d_in[1];
  const int* srcI  = eidx;
  const int* dstI  = eidx + EE;
  const float* ea0 = (const float*)d_in[2];
  const float* u0  = (const float*)d_in[3];
  const int* batch = (const int*)d_in[4];
  const float* edge_w1 = (const float*)d_in[5],  *edge_b1 = (const float*)d_in[6];
  const float* edge_w2 = (const float*)d_in[7],  *edge_b2 = (const float*)d_in[8];
  const float* node_w1 = (const float*)d_in[9],  *node_b1 = (const float*)d_in[10];
  const float* node_w2 = (const float*)d_in[11], *node_b2 = (const float*)d_in[12];
  const float* glob_w1 = (const float*)d_in[13], *glob_b1 = (const float*)d_in[14];
  const float* glob_w2 = (const float*)d_in[15], *glob_b2 = (const float*)d_in[16];
  const float* egru_wih = (const float*)d_in[17], *egru_whh = (const float*)d_in[18];
  const float* egru_bih = (const float*)d_in[19], *egru_bhh = (const float*)d_in[20];
  const float* ngru_wih = (const float*)d_in[21], *ngru_whh = (const float*)d_in[22];
  const float* ngru_bih = (const float*)d_in[23], *ngru_bhh = (const float*)d_in[24];
  const float* ggru_wih = (const float*)d_in[25], *ggru_whh = (const float*)d_in[26];
  const float* ggru_bih = (const float*)d_in[27], *ggru_bhh = (const float*)d_in[28];
  float* outp = (float*)d_out;

  // Workspace carve (~308 MB). All region offsets keep 16B alignment.
  char* w = (char*)d_ws;
  float* ea   = (float*)w; w += (size_t)EE * H * 4;  // edge_attr state
  float* xb   = (float*)w; w += (size_t)NN * H * 4;  // node state
  float* ub   = (float*)w; w += (size_t)BB * H * 4;  // global state
  float* agg  = (float*)w; w += (size_t)NN * H * 4;  // edge->node sums
  float* acnt = (float*)w; w += (size_t)NN * 4;      // edge->node counts
  float* gm   = (float*)w; w += (size_t)BB * H * 4;  // node->graph sums
  float* gcnt = (float*)w; w += (size_t)BB * 4;      // node->graph counts
  _Float16* arena = (_Float16*)w;

  _Float16* ew1  = arena;          // 128 x 512
  _Float16* ew2  = ew1 + 65536;    // 128 x 128
  _Float16* nw1  = ew2 + 16384;    // 128 x 384
  _Float16* nw2  = nw1 + 49152;    // 128 x 128
  _Float16* gw1  = nw2 + 16384;    // 128 x 256
  _Float16* gw2  = gw1 + 32768;    // 128 x 128
  _Float16* ewih = gw2 + 16384;    // 384 x 128 each below
  _Float16* ewhh = ewih + 49152;
  _Float16* nwih = ewhh + 49152;
  _Float16* nwhh = nwih + 49152;
  _Float16* gwih = nwhh + 49152;
  _Float16* gwhh = gwih + 49152;

  auto cvt = [&](const float* s, _Float16* dptr, int n) {
    f32_to_f16_kernel<<<(n + 255) / 256, 256, 0, stream>>>(s, dptr, n);
  };
  cvt(edge_w1, ew1, H * 4 * H);  cvt(edge_w2, ew2, H * H);
  cvt(node_w1, nw1, H * 3 * H);  cvt(node_w2, nw2, H * H);
  cvt(glob_w1, gw1, H * 2 * H);  cvt(glob_w2, gw2, H * H);
  cvt(egru_wih, ewih, 3 * H * H); cvt(egru_whh, ewhh, 3 * H * H);
  cvt(ngru_wih, nwih, 3 * H * H); cvt(ngru_whh, nwhh, 3 * H * H);
  cvt(ggru_wih, gwih, 3 * H * H); cvt(ggru_whh, gwhh, 3 * H * H);

  // Step-invariant segment counts (dst / batch never change).
  zero_kernel<<<(unsigned)((NN + BB + 255) / 256), 256, 0, stream>>>(
      acnt, (size_t)NN);
  zero_kernel<<<1, 256, 0, stream>>>(gcnt, (size_t)BB);
  count_kernel<<<(EE + 255) / 256, 256, 0, stream>>>(dstI, acnt, EE);
  count_kernel<<<(NN + 255) / 256, 256, 0, stream>>>(batch, gcnt, NN);

  const int eTiles = EE / 32;             // 15625
  const int nTiles = (NN + 31) / 32;      // 1563 (tail tile has 16 rows)

  for (int s = 0; s < STEPS; ++s) {
    const float* xs  = (s == 0) ? x0  : xb;
    const float* eas = (s == 0) ? ea0 : ea;
    const float* us  = (s == 0) ? u0  : ub;

    // 1) edge model + edge GRU (in-place on ws after step 0; waves own rows)
    edge_kernel<<<(eTiles + 1) / 2, 64, 0, stream>>>(
        xs, srcI, dstI, batch, eas, ea, us,
        ew1, edge_b1, ew2, edge_b2, ewih, ewhh, egru_bih, egru_bhh, eTiles);

    // 2) segment sum of new edge_attr over dst (mean folded into node kernel)
    zero_kernel<<<(unsigned)(((size_t)NN * H + 255) / 256), 256, 0, stream>>>(
        agg, (size_t)NN * H);
    scatter_edges_kernel<<<(unsigned)(((size_t)EE * H + 255) / 256), 256, 0,
                          stream>>>(ea, dstI, agg);

    // 3) node model + node GRU
    node_kernel<<<(nTiles + 1) / 2, 64, 0, stream>>>(
        xs, xb, agg, acnt, us, batch,
        nw1, node_b1, nw2, node_b2, nwih, nwhh, ngru_bih, ngru_bhh, nTiles);

    // 4) per-graph node sum (mean folded into global kernel)
    zero_kernel<<<(unsigned)(((size_t)BB * H + 255) / 256), 256, 0, stream>>>(
        gm, (size_t)BB * H);
    scatter_nodes_kernel<<<(unsigned)(((size_t)NN * H + 255) / 256), 256, 0,
                          stream>>>(xb, batch, gm);

    // 5) global model + global GRU; also writes d_out[:, s, :]
    global_kernel<<<1, 64, 0, stream>>>(
        us, ub, gm, gcnt, gw1, glob_b1, gw2, glob_b2,
        gwih, gwhh, ggru_bih, ggru_bhh, outp, s);
  }
}
